// MoEGate_24275155157556
// MI455X (gfx1250) — compile-verified, matrix operand order checked by reading, not run on hardware
//
#include <hip/hip_runtime.h>
#include <hip/hip_bf16.h>

// ---------------------------------------------------------------------------
// MoE gate for MI455X (gfx1250, wave32).
//   logits = X[16384,4096]f32 @ W[4096,64]f32 ; softmax; top-8; renorm; x2.5
// Memory-bound: 268 MB of hidden-state reads @ 23.3 TB/s (~11.5us floor).
// f32 WMMA (K=4) would be compute-bound (~300 TF); f16 WMMA (K=32, ~2.5 PF
// dense) keeps us on the HBM roofline, so we convert f32->f16 in registers.
// Weight is pre-transposed/converted to f16 [64][4096] in d_ws (512 KB,
// L2-resident) so B fragments are contiguous global_load_b128 -> no LDS
// staging / no barriers inside the GEMM K-loop.
// ---------------------------------------------------------------------------

typedef __attribute__((ext_vector_type(16))) _Float16 v16h;
typedef __attribute__((ext_vector_type(8)))  _Float16 v8h;
typedef __attribute__((ext_vector_type(8)))  float    v8f;
typedef __attribute__((ext_vector_type(4)))  float    f32x4;

#define HIDDEN      4096
#define NUM_EXPERTS 64
#define TOP_K       8
#define TOKENS      (4 * 4096)
#define ROUTED_SCALING 2.5f
#define WAVES_PER_BLOCK 8
#define TOKENS_PER_WAVE 16

// --- Prepass: W[k][e] f32 -> WT[e][k] f16 (coalesced read, scatter write) ---
__global__ __launch_bounds__(256)
void moe_gate_wcvt_kernel(const float* __restrict__ W, _Float16* __restrict__ WT) {
    int flat = blockIdx.x * 256 + threadIdx.x;   // 0 .. HIDDEN*NUM_EXPERTS-1
    int k = flat >> 6;                           // / NUM_EXPERTS
    int e = flat & (NUM_EXPERTS - 1);
    WT[(size_t)e * HIDDEN + k] = (_Float16)W[flat];
}

// --- Main: GEMM via v_wmma_f32_16x16x32_f16 + softmax/top-8 epilogue -------
__global__ __launch_bounds__(256)
void moe_gate_main_kernel(const float* __restrict__ X,      // [TOKENS][HIDDEN]
                          const _Float16* __restrict__ WT,  // [E][HIDDEN] f16
                          const float* __restrict__ bias_p, // scalar
                          float* __restrict__ out)          // weights | indices
{
    __shared__ float s_logits[WAVES_PER_BLOCK][TOKENS_PER_WAVE][NUM_EXPERTS];

    const int lane = threadIdx.x & 31;           // wave32
    const int wid  = threadIdx.x >> 5;
    const int wave_global = blockIdx.x * WAVES_PER_BLOCK + wid;
    const int m0 = wave_global * TOKENS_PER_WAVE;

    const int mr = lane & 15;   // row (A) / column (B) within 16
    const int hk = lane >> 4;   // K-half selector per ISA 16-bit layout

    const float* __restrict__ arow = X + (size_t)(m0 + mr) * HIDDEN;

    v8f c0 = {}, c1 = {}, c2 = {}, c3 = {};

    for (int kk = 0; kk < HIDDEN; kk += 32) {
        // A fragment 16x32: ISA layout -> lane holds K [hk*8..hk*8+7] in
        // VGPR0-3 (elts 0..7) and K [16+hk*8..+7] in VGPR4-7 (elts 8..15).
        const int ka = kk + hk * 8;
        f32x4 af0 = *(const f32x4*)(arow + ka);
        f32x4 af1 = *(const f32x4*)(arow + ka + 4);
        f32x4 af2 = *(const f32x4*)(arow + ka + 16);
        f32x4 af3 = *(const f32x4*)(arow + ka + 20);
        v16h a;
        #pragma unroll
        for (int i = 0; i < 4; ++i) {
            a[i]      = (_Float16)af0[i];
            a[4 + i]  = (_Float16)af1[i];
            a[8 + i]  = (_Float16)af2[i];
            a[12 + i] = (_Float16)af3[i];
        }

        // B fragments 32x16 (column-major mirror of A layout): lane holds
        // column n = t*16+mr, same K packing. WT rows are K-contiguous f16,
        // so each half-fragment is one 16-byte global_load_b128 from L2.
        v16h b0v, b1v, b2v, b3v;
        {
            const _Float16* br;
            br = WT + (size_t)(0 * 16 + mr) * HIDDEN + ka;
            v8h lo = *(const v8h*)br, hi = *(const v8h*)(br + 16);
            #pragma unroll
            for (int i = 0; i < 8; ++i) { b0v[i] = lo[i]; b0v[8 + i] = hi[i]; }
            br = WT + (size_t)(1 * 16 + mr) * HIDDEN + ka;
            lo = *(const v8h*)br; hi = *(const v8h*)(br + 16);
            #pragma unroll
            for (int i = 0; i < 8; ++i) { b1v[i] = lo[i]; b1v[8 + i] = hi[i]; }
            br = WT + (size_t)(2 * 16 + mr) * HIDDEN + ka;
            lo = *(const v8h*)br; hi = *(const v8h*)(br + 16);
            #pragma unroll
            for (int i = 0; i < 8; ++i) { b2v[i] = lo[i]; b2v[8 + i] = hi[i]; }
            br = WT + (size_t)(3 * 16 + mr) * HIDDEN + ka;
            lo = *(const v8h*)br; hi = *(const v8h*)(br + 16);
            #pragma unroll
            for (int i = 0; i < 8; ++i) { b3v[i] = lo[i]; b3v[8 + i] = hi[i]; }
        }

        // D = A x B + C   (EXEC is all-ones here; divergence only after loop)
        c0 = __builtin_amdgcn_wmma_f32_16x16x32_f16(false, a, false, b0v, (short)0, c0, false, false);
        c1 = __builtin_amdgcn_wmma_f32_16x16x32_f16(false, a, false, b1v, (short)0, c1, false, false);
        c2 = __builtin_amdgcn_wmma_f32_16x16x32_f16(false, a, false, b2v, (short)0, c2, false, false);
        c3 = __builtin_amdgcn_wmma_f32_16x16x32_f16(false, a, false, b3v, (short)0, c3, false, false);
    }

    // C layout: lane (n=mr, half=hk), VGPR r -> element (m = r + 8*hk, n).
    #pragma unroll
    for (int r = 0; r < 8; ++r) {
        const int mloc = r + 8 * hk;
        s_logits[wid][mloc][ 0 + mr] = c0[r];
        s_logits[wid][mloc][16 + mr] = c1[r];
        s_logits[wid][mloc][32 + mr] = c2[r];
        s_logits[wid][mloc][48 + mr] = c3[r];
    }
    __syncthreads();

    // One lane per token: softmax over 64 experts, top-8 with
    // lowest-index-wins tie-breaking (matches jax.lax.top_k).
    if (lane < TOKENS_PER_WAVE) {
        const float bias = bias_p[0];
        const float* lg = s_logits[wid][lane];
        const int tok = m0 + lane;

        float mx = -__builtin_inff();
        for (int e = 0; e < NUM_EXPERTS; ++e) mx = fmaxf(mx, lg[e]);
        float denom = 0.0f;
        for (int e = 0; e < NUM_EXPERTS; ++e) denom += __expf(lg[e] - mx);
        const float inv = 1.0f / denom;

        float tw[TOP_K]; int ti[TOP_K];
        #pragma unroll
        for (int j = 0; j < TOP_K; ++j) { tw[j] = -__builtin_inff(); ti[j] = 0; }
        for (int e = 0; e < NUM_EXPERTS; ++e) {
            const float p = __expf(lg[e] - mx) * inv;   // softmax score
            const float cv = p + bias;                  // corrected score
            if (cv > tw[TOP_K - 1]) {                   // strict > : stable ties
                int j = TOP_K - 1;
                while (j > 0 && cv > tw[j - 1]) {
                    tw[j] = tw[j - 1]; ti[j] = ti[j - 1]; --j;
                }
                tw[j] = cv; ti[j] = e;
            }
        }
        float s = 0.0f;
        #pragma unroll
        for (int j = 0; j < TOP_K; ++j) s += (tw[j] - bias);   // raw probs
        const float norm = ROUTED_SCALING / (s + 1e-20f);
        #pragma unroll
        for (int j = 0; j < TOP_K; ++j) {
            out[(size_t)tok * TOP_K + j] = (tw[j] - bias) * norm;
            // tuple concat (f32, i32) promotes to f32 -> store index as float
            out[(size_t)TOKENS * TOP_K + (size_t)tok * TOP_K + j] = (float)ti[j];
        }
    }
}

extern "C" void kernel_launch(void* const* d_in, const int* in_sizes, int n_in,
                              void* d_out, int out_size, void* d_ws, size_t ws_size,
                              hipStream_t stream) {
    const float* X    = (const float*)d_in[0];   // [4,4096,4096] f32
    const float* W    = (const float*)d_in[1];   // [4096,64] f32
    const float* bias = (const float*)d_in[2];   // scalar f32
    float* out        = (float*)d_out;
    _Float16* WT      = (_Float16*)d_ws;         // needs 4096*64*2 = 512 KB

    // 1) transpose+convert router weight to f16 (one-shot, L2-resident)
    moe_gate_wcvt_kernel<<<(HIDDEN * NUM_EXPERTS) / 256, 256, 0, stream>>>(W, WT);

    // 2) fused gate GEMM (WMMA f16) + softmax + top-8
    const int blocks = TOKENS / (WAVES_PER_BLOCK * TOKENS_PER_WAVE); // 128
    moe_gate_main_kernel<<<blocks, 256, 0, stream>>>(X, WT, bias, out);
}
